// EWMAEmbedding_16784732193306
// MI455X (gfx1250) — compile-verified, matrix operand order checked by reading, not run on hardware
//
#include <hip/hip_runtime.h>

// ---------------------------------------------------------------------------
// EWMAEmbedding for MI455X (gfx1250, wave32)
//   x:[32,512,2048] f32, W:[1024,512] f32, b:[1024] f32 -> out:[32,2048,1024] f32
//   Phase 0: W f32 -> bf16 (1 MB, workspace)
//   Phase 1: segmented bidirectional EWMA -> bf16 A[M=65536, K=512] (workspace)
//   Phase 2: bf16 WMMA GEMM, double-buffered LDS fed by
//            global_load_async_to_lds_b128 (ASYNCcnt), + bias -> out
// ---------------------------------------------------------------------------

#define B_DIM   32
#define C_IN    512
#define T_DIM   2048
#define D_MODEL 1024
#define M_DIM   (B_DIM * T_DIM)   // 65536
#define ALPHA_F 0.1f
#define OMA_F   0.9f
#define SEG     512               // T segment length
#define WARM    256               // scan warm-up (0.9^256 ~ 2e-12)

typedef __bf16 v16bf __attribute__((ext_vector_type(16)));
typedef float  v8f   __attribute__((ext_vector_type(8)));
typedef unsigned int uint4v __attribute__((ext_vector_type(4)));

__device__ __forceinline__ unsigned short f32_to_bf16(float f) {
    union { float f; unsigned u; } v; v.f = f;
    unsigned r = v.u + 0x7FFFu + ((v.u >> 16) & 1u);   // round-to-nearest-even
    return (unsigned short)(r >> 16);
}
__device__ __forceinline__ float bf16_to_f32(unsigned short h) {
    union { unsigned u; float f; } v; v.u = ((unsigned)h) << 16;
    return v.f;
}
// low 32 bits of a flat pointer into LDS = LDS byte offset (aperture rule)
__device__ __forceinline__ unsigned lds_off(const void* p) {
    return (unsigned)(unsigned long long)p;
}
__device__ __forceinline__ void async_b128(unsigned ldsdst, const void* gsrc) {
    asm volatile("global_load_async_to_lds_b128 %0, %1, off"
                 :: "v"(ldsdst), "v"(gsrc) : "memory");
}
__device__ __forceinline__ void wait_async0() {
    asm volatile("s_wait_asynccnt 0x0" ::: "memory");
}

// ---------------------------------------------------------------------------
// Phase 0: convert W [1024,512] f32 -> bf16. 256 blocks x 256 thr x 8 elems.
// ---------------------------------------------------------------------------
__global__ __launch_bounds__(256)
void convert_w_kernel(const float* __restrict__ W, unsigned short* __restrict__ Wb)
{
    const int i = (blockIdx.x * 256 + threadIdx.x) * 8;
    union { uint4v u; float f[4]; } lo, hi;
    lo.u = ((const uint4v*)(W + i))[0];
    hi.u = ((const uint4v*)(W + i))[1];
    union { unsigned short s[8]; uint4v u; } p;
    #pragma unroll
    for (int e = 0; e < 4; ++e) p.s[e]     = f32_to_bf16(lo.f[e]);
    #pragma unroll
    for (int e = 0; e < 4; ++e) p.s[4 + e] = f32_to_bf16(hi.f[e]);
    *(uint4v*)(Wb + i) = p.u;
}

// ---------------------------------------------------------------------------
// Phase 1: segmented bidirectional EWMA.
// 256 blocks x 8 waves = 2048 waves. Block -> (b, segment, half of c-groups);
// all waves in a block share the segment -> uniform __syncthreads counts.
// Wave: 32 c-lanes; x tiles staged via LDS (coalesced along T, conflict-free
// pitch-33 column reads); stores along C are contiguous (A is [M,K] row-major).
// ---------------------------------------------------------------------------
__global__ __launch_bounds__(256)
void ewma_bidir_kernel(const float* __restrict__ x, unsigned short* __restrict__ Amat)
{
    __shared__ float tile[8][32][33];

    const int tid  = threadIdx.x;
    const int w    = tid >> 5;
    const int lane = tid & 31;

    const int combo = blockIdx.x >> 1;        // 0..127 : (b, s)
    const int half  = blockIdx.x & 1;         // 0..1
    const int b     = combo >> 2;             // 0..31
    const int s     = combo & 3;              // 0..3 segment
    const int c0    = (half * 8 + w) * 32;    // 0..480

    const size_t xbase = ((size_t)b * C_IN + c0) * T_DIM;   // + r*T + t
    const size_t abase = (size_t)b * T_DIM * C_IN + c0;     // + t*C + lane

    const int tlo = s * SEG;                  // first stored t
    const int thi = tlo + SEG;                // end of stored (exclusive)

    // ---- backward: g_t = a*x_t + (1-a)*g_{t+1}, warm-up above the segment ----
    {
        const int wend = (s == 3) ? T_DIM : thi + WARM;   // exclusive scan top
        float f = 0.0f;
        for (int t0 = wend - 32; t0 >= tlo; t0 -= 32) {
            #pragma unroll 8
            for (int r = 0; r < 32; ++r)
                tile[w][r][lane] = x[xbase + (size_t)r * T_DIM + t0 + lane];
            __syncthreads();
            const bool storech = (t0 < thi);              // chunk fully stored?
            #pragma unroll
            for (int k = 31; k >= 0; --k) {
                const int t = t0 + k;
                const float xv = tile[w][lane][k];
                f = (t == wend - 1) ? xv : (ALPHA_F * xv + OMA_F * f);
                if (storech)
                    Amat[abase + (size_t)t * C_IN + lane] = f32_to_bf16(0.5f * f);
            }
            __syncthreads();
        }
    }

    // ---- forward: f_t = a*x_t + (1-a)*f_{t-1}, warm-up below; combine ----
    {
        const int wstart = (s == 0) ? 0 : tlo - WARM;
        float f = 0.0f;
        for (int t0 = wstart; t0 < thi; t0 += 32) {
            #pragma unroll 8
            for (int r = 0; r < 32; ++r)
                tile[w][r][lane] = x[xbase + (size_t)r * T_DIM + t0 + lane];
            __syncthreads();
            const bool storech = (t0 >= tlo);
            #pragma unroll
            for (int k = 0; k < 32; ++k) {
                const int t = t0 + k;
                const float xv = tile[w][lane][k];
                f = (t == wstart) ? xv : (ALPHA_F * xv + OMA_F * f);
                if (storech) {
                    const size_t idx = abase + (size_t)t * C_IN + lane;
                    const float prev = bf16_to_f32(Amat[idx]);  // same lane wrote it
                    Amat[idx] = f32_to_bf16(0.5f * f + prev);
                }
            }
            __syncthreads();
        }
    }
}

// ---------------------------------------------------------------------------
// Phase 2: GEMM  out[m,n] = sum_k A[m,k]*Wb[n,k] + bias[n]
// Block tile 128x128, K-chunks of 64, double-buffered LDS filled by
// global_load_async_to_lds_b128. 8 waves (wm 0..3, wn 0..1) each compute
// 32x64 via 2x4 grid of v_wmma_f32_16x16x32_bf16.
// bf16 operand swizzle (CDNA5 ISA): lane = 16*h + r holds row r,
// K = {h*8..h*8+7} U {16+h*8..+7}: two contiguous 16B LDS chunks per frag.
// ---------------------------------------------------------------------------
__device__ __forceinline__ v16bf load_frag(const unsigned short* rowp, int kb, int lh)
{
    union { uint4v u[2]; v16bf v; } fr;
    fr.u[0] = *(const uint4v*)(rowp + kb + lh * 8);
    fr.u[1] = *(const uint4v*)(rowp + kb + lh * 8 + 16);
    return fr.v;
}

__global__ __launch_bounds__(256)
void wmma_gemm_bias_kernel(const unsigned short* __restrict__ Amat,
                           const unsigned short* __restrict__ Wb,
                           const float* __restrict__ bias,
                           float* __restrict__ out)
{
    __shared__ __align__(16) unsigned short lds_a[2][128][72];  // pitch 144B
    __shared__ __align__(16) unsigned short lds_b[2][128][72];

    const int tid  = threadIdx.x;
    const int bm   = blockIdx.y;            // 0..511
    const int bn   = blockIdx.x;            // 0..7
    const int wave = tid >> 5;
    const int lane = tid & 31;
    const int wm   = wave >> 1;             // 0..3 -> 32 rows
    const int wn   = wave & 1;              // 0..1 -> 64 cols
    const int lr   = lane & 15;
    const int lh   = lane >> 4;

    const size_t a_row0 = (size_t)bm * 128;
    const size_t w_row0 = (size_t)bn * 128;

    // per-thread async copy slots: 4 chunks of 16B for A, 4 for B
    // chunk c = tid + 256*j -> row = c>>3 (0..127), ko = (c&7)*8
    auto issue_tile = [&](int buf, int kc) {
        #pragma unroll
        for (int j = 0; j < 4; ++j) {
            const int c   = tid + 256 * j;
            const int row = c >> 3;
            const int ko  = (c & 7) * 8;
            async_b128(lds_off(&lds_a[buf][row][ko]),
                       Amat + (a_row0 + row) * C_IN + kc * 64 + ko);
            async_b128(lds_off(&lds_b[buf][row][ko]),
                       Wb + (w_row0 + row) * C_IN + kc * 64 + ko);
        }
    };

    v8f acc[2][4];
    #pragma unroll
    for (int tm = 0; tm < 2; ++tm)
        #pragma unroll
        for (int tn = 0; tn < 4; ++tn)
            acc[tm][tn] = (v8f)0.0f;

    issue_tile(0, 0);
    wait_async0();
    __syncthreads();

    for (int kc = 0; kc < C_IN / 64; ++kc) {              // 8 chunks of K=64
        const int cur = kc & 1;
        if (kc + 1 < C_IN / 64)
            issue_tile(cur ^ 1, kc + 1);                  // overlap next fill
        if (kc + 2 < C_IN / 64)                           // L2 prefetch, chunk+2
            __builtin_prefetch(Amat + (a_row0 + (tid >> 1)) * C_IN +
                               (kc + 2) * 64 + (tid & 1) * 32, 0, 1);

        #pragma unroll
        for (int ks = 0; ks < 2; ++ks) {                  // 2 K-steps of 32
            const int kb = ks * 32;
            v16bf afrag[2], bfrag[4];
            #pragma unroll
            for (int tm = 0; tm < 2; ++tm)
                afrag[tm] = load_frag(&lds_a[cur][wm * 32 + tm * 16 + lr][0], kb, lh);
            #pragma unroll
            for (int tn = 0; tn < 4; ++tn)
                bfrag[tn] = load_frag(&lds_b[cur][wn * 64 + tn * 16 + lr][0], kb, lh);
            #pragma unroll
            for (int tm = 0; tm < 2; ++tm)
                #pragma unroll
                for (int tn = 0; tn < 4; ++tn)
                    acc[tm][tn] = __builtin_amdgcn_wmma_f32_16x16x32_bf16(
                        false, afrag[tm], false, bfrag[tn],
                        (short)0, acc[tm][tn], false, false);
        }

        if (kc + 1 < C_IN / 64)
            wait_async0();          // next buffer fully in LDS
        __syncthreads();            // + all waves done reading cur
    }

    // ---- epilogue: D layout: VGPR i -> (M = i + 8*lh, N = lr); + bias ----
    #pragma unroll
    for (int tn = 0; tn < 4; ++tn) {
        const int col = bn * 128 + wn * 64 + tn * 16 + lr;
        const float bv = bias[col];
        #pragma unroll
        for (int tm = 0; tm < 2; ++tm) {
            const int row0 = bm * 128 + wm * 32 + tm * 16 + lh * 8;
            #pragma unroll
            for (int i = 0; i < 8; ++i)
                out[(size_t)(row0 + i) * D_MODEL + col] = acc[tm][tn][i] + bv;
        }
    }
}

// ---------------------------------------------------------------------------
extern "C" void kernel_launch(void* const* d_in, const int* in_sizes, int n_in,
                              void* d_out, int out_size, void* d_ws, size_t ws_size,
                              hipStream_t stream)
{
    const float* x    = (const float*)d_in[0];   // [32,512,2048]
    const float* W    = (const float*)d_in[1];   // [1024,512]
    const float* bias = (const float*)d_in[2];   // [1024]
    float* out        = (float*)d_out;           // [32,2048,1024]

    unsigned short* Amat = (unsigned short*)d_ws;                     // 64 MB bf16
    unsigned short* Wb   = (unsigned short*)((char*)d_ws +
                            (size_t)M_DIM * C_IN * sizeof(unsigned short)); // +1 MB

    convert_w_kernel<<<dim3((D_MODEL * C_IN) / (256 * 8)), dim3(256), 0, stream>>>(W, Wb);
    ewma_bidir_kernel<<<dim3(256), dim3(256), 0, stream>>>(x, Amat);

    dim3 grid(D_MODEL / 128, M_DIM / 128);       // (8, 512)
    wmma_gemm_bias_kernel<<<grid, dim3(256), 0, stream>>>(Amat, Wb, bias, out);
}
